// Layer_81458349736170
// MI455X (gfx1250) — matrix operand order repaired and checked by reading, not compiled, not run on hardware
//
#include <hip/hip_runtime.h>
#include <hip/hip_bf16.h>
#include <math.h>

typedef __attribute__((ext_vector_type(16))) __bf16 bf16x16;
typedef __attribute__((ext_vector_type(8)))  __bf16 bf16x8;
typedef __attribute__((ext_vector_type(8)))  float  f32x8;

#define LN_EPSF 1e-5f
#define BB 2
#define SS 2048
#define DD 2048
#define HH 16
#define HDIM 128
#define FF 8192
#define BS (BB*SS)

static __device__ __forceinline__ __bf16 f2bf(float x) { return (__bf16)x; }

static __device__ __forceinline__ bf16x16 cat8(bf16x8 lo, bf16x8 hi) {
  return __builtin_shufflevector(lo, hi, 0,1,2,3,4,5,6,7,8,9,10,11,12,13,14,15);
}

// Per-lane async copy of 16 bytes global -> LDS (ASYNCcnt-tracked, no VGPR data).
static __device__ __forceinline__ void async_cp16(const void* lds, const void* gaddr) {
  unsigned l = (unsigned)(uintptr_t)lds;                 // low 32 bits = LDS offset
  unsigned long long g = (unsigned long long)(uintptr_t)gaddr;
  asm volatile("global_load_async_to_lds_b128 %0, %1, off" :: "v"(l), "v"(g) : "memory");
}
static __device__ __forceinline__ void async_wait0() {
  asm volatile("s_wait_asynccnt 0x0" ::: "memory");
}

// ---------------------------------------------------------------------------
// float -> bf16 elementwise
// ---------------------------------------------------------------------------
__global__ void cvt_bf16(const float* __restrict__ s, __bf16* __restrict__ d, size_t n) {
  size_t i = (size_t)blockIdx.x * 256 + threadIdx.x;
  if (i < n) d[i] = f2bf(s[i]);
}

// ---------------------------------------------------------------------------
// W[K][N] f32  ->  WT[N][K] bf16   (32x32 LDS tiled transpose)
// ---------------------------------------------------------------------------
__global__ __launch_bounds__(256)
void transpose_cvt(const float* __restrict__ W, __bf16* __restrict__ WT, int K, int N) {
  __shared__ float t[32][33];
  int bx = blockIdx.x * 32;            // n
  int by = blockIdx.y * 32;            // k
  int tx = threadIdx.x & 31, ty = threadIdx.x >> 5;
  for (int i = 0; i < 4; i++)
    t[ty + i * 8][tx] = W[(size_t)(by + ty + i * 8) * N + bx + tx];
  __syncthreads();
  for (int i = 0; i < 4; i++)
    WT[(size_t)(bx + ty + i * 8) * K + by + tx] = f2bf(t[tx][ty + i * 8]);
}

// ---------------------------------------------------------------------------
// V pack: VF [B,S,D] f32 -> vt [B,H,hd,S] bf16, coalesced tiled transpose.
// ---------------------------------------------------------------------------
__global__ __launch_bounds__(256)
void pack_vT(const float* __restrict__ VF, __bf16* __restrict__ vt) {
  __shared__ float t[32][33];
  int bh = blockIdx.z;                 // b*H + h
  int b = bh / HH, h = bh % HH;
  int s0 = blockIdx.y * 32;
  int d0 = blockIdx.x * 32;            // within head
  int tx = threadIdx.x & 31, ty = threadIdx.x >> 5;
  for (int i = 0; i < 4; i++) {
    int s = s0 + ty + i * 8;
    t[ty + i * 8][tx] = VF[((size_t)b * SS + s) * DD + h * HDIM + d0 + tx];
  }
  __syncthreads();
  for (int i = 0; i < 4; i++) {
    int d = d0 + ty + i * 8;
    vt[((size_t)bh * HDIM + d) * SS + s0 + tx] = f2bf(t[tx][ty + i * 8]);
  }
}

// ---------------------------------------------------------------------------
// GEMM: C[MxN] = A[MxK] @ BT[NxK]^T, bf16 inputs, f32 WMMA accumulate.
// 128x128 block, 8 waves, wave tile 32x64, K-step 64 (2 WMMA chunks/stage),
// double-buffered LDS, all staging via global_load_async_to_lds_b128.
// EPI==0: store f32 C.  EPI==1: store bf16 gelu(C + bias[col]) to Cb.
// ---------------------------------------------------------------------------
#define BLK_M 128
#define BLK_N 128
#define BLK_K 64
#define KPAD  8
#define LDT   (BLK_K + KPAD)   // 72 elems = 144B rows (16B aligned)

template <int EPI>
__global__ __launch_bounds__(256)
void gemm_bf16_async(const __bf16* __restrict__ A, const __bf16* __restrict__ BT,
                     float* __restrict__ C, int M, int N, int K,
                     const float* __restrict__ bias, __bf16* __restrict__ Cb) {
  __shared__ __bf16 As[2][BLK_M][LDT];
  __shared__ __bf16 Bs[2][BLK_N][LDT];   // [n][k]
  const int tid  = threadIdx.x;
  const int wid  = tid >> 5;
  const int lane = tid & 31;
  const int l16  = lane & 15;
  const int lh   = lane >> 4;
  const int bm = blockIdx.y * BLK_M;
  const int bn = blockIdx.x * BLK_N;
  const int wm = (wid & 3) * 32;
  const int wn = (wid >> 2) * 64;

  f32x8 acc[2][4];
  for (int i = 0; i < 2; i++)
    for (int j = 0; j < 4; j++)
      for (int v = 0; v < 8; v++) acc[i][j][v] = 0.0f;

  auto stage = [&](int buf, int k0) {
    // A tile: 128 rows x 64 bf16 = 1024 chunks of 16B; 4 per thread
    for (int p = 0; p < 4; p++) {
      int idx = p * 256 + tid;
      int r = idx >> 3, c8 = (idx & 7) * 8;
      async_cp16(&As[buf][r][c8], A + (size_t)(bm + r) * K + k0 + c8);
    }
    // BT tile: 128 rows x 64 bf16; 4 per thread
    for (int p = 0; p < 4; p++) {
      int idx = p * 256 + tid;
      int r = idx >> 3, c8 = (idx & 7) * 8;
      async_cp16(&Bs[buf][r][c8], BT + (size_t)(bn + r) * K + k0 + c8);
    }
  };

  stage(0, 0);
  async_wait0();
  __syncthreads();
  const int nk = K / BLK_K;
  for (int kt = 0; kt < nk; kt++) {
    int buf = kt & 1;
    if (kt + 1 < nk) stage(buf ^ 1, (kt + 1) * BLK_K);
    for (int kc = 0; kc < 2; kc++) {
      bf16x16 af[2];
      for (int i = 0; i < 2; i++) {
        int r  = wm + i * 16 + l16;
        int ka = kc * 32 + (lh ? 8 : 0);     // A-frag: K chunks {0,16} / {8,24}
        af[i] = cat8(*(const bf16x8*)&As[buf][r][ka],
                     *(const bf16x8*)&As[buf][r][ka + 16]);
      }
      bf16x16 bfr[4];
      for (int j = 0; j < 4; j++) {
        int c  = wn + j * 16 + l16;
        int kb = kc * 32 + (lh ? 8 : 0);     // B-frag: K chunks {0,16} / {8,24}
        bfr[j] = cat8(*(const bf16x8*)&Bs[buf][c][kb],
                      *(const bf16x8*)&Bs[buf][c][kb + 16]);
      }
      for (int i = 0; i < 2; i++)
        for (int j = 0; j < 4; j++)
          acc[i][j] = __builtin_amdgcn_wmma_f32_16x16x32_bf16(
              false, af[i], false, bfr[j], (short)0, acc[i][j], false, false);
    }
    async_wait0();
    __syncthreads();
  }
  for (int i = 0; i < 2; i++) {
    int rb = bm + wm + i * 16 + (lh ? 8 : 0);
    for (int j = 0; j < 4; j++) {
      int cb = bn + wn + j * 16 + l16;
      for (int v = 0; v < 8; v++) {
        if (EPI == 0) {
          C[(size_t)(rb + v) * N + cb] = acc[i][j][v];
        } else {
          float t = acc[i][j][v] + bias[cb];
          t = 0.5f * t * (1.0f + erff(t * 0.70710678118654752f));
          Cb[(size_t)(rb + v) * N + cb] = f2bf(t);
        }
      }
    }
  }
}

// ---------------------------------------------------------------------------
// RoPE (reference indexes sin/cos by BATCH: L = x.shape[0]) + pack:
// q,k -> bf16 [B,H,S,hd] (coalesced within head).
// ---------------------------------------------------------------------------
__global__ void rope_pack(const float* __restrict__ qf, const float* __restrict__ kf,
                          const float* __restrict__ sn, const float* __restrict__ cs,
                          __bf16* __restrict__ qb, __bf16* __restrict__ kb) {
  size_t t = (size_t)blockIdx.x * 256 + threadIdx.x;
  if (t >= (size_t)BB * SS * (DD / 2)) return;
  int j = (int)(t % (DD / 2));
  size_t bs = t / (DD / 2);
  int s = (int)(bs % SS);
  int b = (int)(bs / SS);
  size_t ib = bs * DD;
  float c  = cs[(size_t)b * (DD / 2) + j];
  float si = sn[(size_t)b * (DD / 2) + j];
  auto putQK = [&](__bf16* dst, int d, float val) {
    int hh = d >> 7, dd = d & 127;
    dst[(((size_t)b * HH + hh) * SS + s) * HDIM + dd] = f2bf(val);
  };
  float x1 = qf[ib + 2 * j], x2 = qf[ib + 2 * j + 1];
  putQK(qb, j, x1 * c - x2 * si);
  putQK(qb, j + DD / 2, x1 * si + x2 * c);
  x1 = kf[ib + 2 * j]; x2 = kf[ib + 2 * j + 1];
  putQK(kb, j, x1 * c - x2 * si);
  putQK(kb, j + DD / 2, x1 * si + x2 * c);
}

// ---------------------------------------------------------------------------
// Flash attention (causal). 64 q-rows / block, 4 waves, 32-wide KV tiles.
// Q,K: bf16 [B,H,S,hd]; Vt: bf16 [B,H,hd,S]; O: f32 in [B,S,D] layout.
// All tile staging via async LDS copies.
// ---------------------------------------------------------------------------
#define FA_Q 64
#define FA_K 32
#define FPAD 8

__global__ __launch_bounds__(128)
void flash_attn(const __bf16* __restrict__ Q, const __bf16* __restrict__ Kt,
                const __bf16* __restrict__ Vt, float* __restrict__ O) {
  __shared__ __bf16 Qs[FA_Q][HDIM + FPAD];
  __shared__ __bf16 Ks[FA_K][HDIM + FPAD];      // [kv][hd] == B-layout for QK^T
  __shared__ __bf16 VT[HDIM][FA_K + FPAD];      // [hd][kv] == B-layout for PV
  __shared__ __bf16 Ps[FA_Q][FA_K + FPAD];      // P staged to A-layout
  const int tid = threadIdx.x, wid = tid >> 5, lane = tid & 31;
  const int l16 = lane & 15, lh = lane >> 4;
  const int blk = blockIdx.x;
  const int qb = (blk % (SS / FA_Q)) * FA_Q;
  const int h  = (blk / (SS / FA_Q)) % HH;
  const int b  = blk / ((SS / FA_Q) * HH);
  const __bf16* Qb_ = Q  + (((size_t)b * HH + h) * SS) * HDIM;
  const __bf16* Kb_ = Kt + (((size_t)b * HH + h) * SS) * HDIM;
  const __bf16* Vb_ = Vt + (((size_t)b * HH + h) * HDIM) * SS;

  // stage Q tile: 64x128 bf16 = 1024 16B-chunks, 8/thread
  for (int p = 0; p < 8; p++) {
    int idx = p * 128 + tid;
    int r = idx >> 4, c8 = (idx & 15) * 8;
    async_cp16(&Qs[r][c8], Qb_ + (size_t)(qb + r) * HDIM + c8);
  }
  async_wait0();
  __syncthreads();
  bf16x16 qfrag[4];
  {
    int r = wid * 16 + l16;
    for (int kc = 0; kc < 4; kc++) {
      int ka = kc * 32 + (lh ? 8 : 0);
      qfrag[kc] = cat8(*(const bf16x8*)&Qs[r][ka], *(const bf16x8*)&Qs[r][ka + 16]);
    }
  }

  f32x8 oacc[8];
  for (int n = 0; n < 8; n++) for (int v = 0; v < 8; v++) oacc[n][v] = 0.0f;
  float mrow[8], lrow[8];
  for (int v = 0; v < 8; v++) { mrow[v] = -INFINITY; lrow[v] = 0.0f; }
  const float scale = 0.08838834764831845f;   // 1/sqrt(128)

  for (int kv = 0; kv < qb + FA_Q; kv += FA_K) {
    __syncthreads();   // previous iteration's LDS reads complete
    // K tile: 32x128 = 512 chunks, 4/thread
    for (int p = 0; p < 4; p++) {
      int idx = p * 128 + tid;
      int r = idx >> 4, c8 = (idx & 15) * 8;
      async_cp16(&Ks[r][c8], Kb_ + (size_t)(kv + r) * HDIM + c8);
    }
    // V^T tile: 128 rows x 32 kv = 512 chunks, 4/thread
    for (int p = 0; p < 4; p++) {
      int idx = p * 128 + tid;
      int r = idx >> 2, c8 = (idx & 3) * 8;
      async_cp16(&VT[r][c8], Vb_ + (size_t)r * SS + kv + c8);
    }
    async_wait0();
    __syncthreads();

    // scores: S = Q @ K^T
    f32x8 sfr[2];
    for (int j = 0; j < 2; j++) for (int v = 0; v < 8; v++) sfr[j][v] = 0.0f;
    for (int kc = 0; kc < 4; kc++) {
      bf16x16 b0, b1;
      {
        int kb2 = kc * 32 + (lh ? 8 : 0);
        b0 = cat8(*(const bf16x8*)&Ks[l16][kb2], *(const bf16x8*)&Ks[l16][kb2 + 16]);
        b1 = cat8(*(const bf16x8*)&Ks[16 + l16][kb2], *(const bf16x8*)&Ks[16 + l16][kb2 + 16]);
      }
      sfr[0] = __builtin_amdgcn_wmma_f32_16x16x32_bf16(
          false, qfrag[kc], false, b0, (short)0, sfr[0], false, false);
      sfr[1] = __builtin_amdgcn_wmma_f32_16x16x32_bf16(
          false, qfrag[kc], false, b1, (short)0, sfr[1], false, false);
    }

    // scale + causal mask + online softmax
    float rmax[8];
    for (int v = 0; v < 8; v++) {
      int row = qb + wid * 16 + (lh ? 8 : 0) + v;
      float a = -INFINITY;
      for (int j = 0; j < 2; j++) {
        int col = kv + j * 16 + l16;
        float sv = sfr[j][v] * scale;
        if (col > row) sv = -INFINITY;
        sfr[j][v] = sv;
        a = fmaxf(a, sv);
      }
      rmax[v] = a;
    }
    for (int m = 1; m < 16; m <<= 1)
      for (int v = 0; v < 8; v++)
        rmax[v] = fmaxf(rmax[v], __shfl_xor(rmax[v], m, 32));

    float alpha[8], mnew[8];
    for (int v = 0; v < 8; v++) {
      mnew[v]  = fmaxf(mrow[v], rmax[v]);
      alpha[v] = __expf(mrow[v] - mnew[v]);
      mrow[v]  = mnew[v];
    }
    float rsum[8];
    for (int v = 0; v < 8; v++) {
      float ssum = 0.0f;
      for (int j = 0; j < 2; j++) {
        float p = __expf(sfr[j][v] - mnew[v]);
        sfr[j][v] = p;
        ssum += p;
      }
      rsum[v] = ssum;
    }
    for (int m = 1; m < 16; m <<= 1)
      for (int v = 0; v < 8; v++) rsum[v] += __shfl_xor(rsum[v], m, 32);
    for (int v = 0; v < 8; v++) {
      lrow[v] = lrow[v] * alpha[v] + rsum[v];
      for (int n = 0; n < 8; n++) oacc[n][v] *= alpha[v];
    }

    // P (C-layout) -> LDS -> A-layout fragment
    for (int v = 0; v < 8; v++) {
      int rr = wid * 16 + (lh ? 8 : 0) + v;
      for (int j = 0; j < 2; j++)
        Ps[rr][j * 16 + l16] = f2bf(sfr[j][v]);
    }
    __syncthreads();

    {
      int r  = wid * 16 + l16;
      int ka = lh ? 8 : 0;
      bf16x16 pf = cat8(*(const bf16x8*)&Ps[r][ka], *(const bf16x8*)&Ps[r][ka + 16]);
      for (int n = 0; n < 8; n++) {
        int c   = n * 16 + l16;
        int kb2 = lh ? 8 : 0;
        bf16x16 vfrag = cat8(*(const bf16x8*)&VT[c][kb2], *(const bf16x8*)&VT[c][kb2 + 16]);
        oacc[n] = __builtin_amdgcn_wmma_f32_16x16x32_bf16(
            false, pf, false, vfrag, (short)0, oacc[n], false, false);
      }
    }
  }

  for (int v = 0; v < 8; v++) {
    float inv = 1.0f / lrow[v];
    int row = qb + wid * 16 + (lh ? 8 : 0) + v;
    for (int n = 0; n < 8; n++) {
      int col = h * HDIM + n * 16 + l16;
      O[((size_t)b * SS + row) * DD + col] = oacc[n][v] * inv;
    }
  }
}

// ---------------------------------------------------------------------------
// out = LayerNorm(a + r [+ bias]); optional bf16 copy of the result.
// ---------------------------------------------------------------------------
__global__ __launch_bounds__(256)
void add_ln(const float* __restrict__ a, const float* __restrict__ r,
            const float* __restrict__ bias, int hasBias,
            float* __restrict__ out, __bf16* __restrict__ outb) {
  __shared__ float red[8];
  const int row = blockIdx.x;
  const int tid = threadIdx.x, wid = tid >> 5, lane = tid & 31;
  const float* pa = a + (size_t)row * DD;
  const float* pr = r + (size_t)row * DD;
  float vals[8];
  float s = 0.0f;
  for (int i = 0; i < 8; i++) {
    int c = i * 256 + tid;
    float v = pa[c] + pr[c] + (hasBias ? bias[c] : 0.0f);
    vals[i] = v; s += v;
  }
  for (int m = 1; m < 32; m <<= 1) s += __shfl_xor(s, m, 32);
  if (lane == 0) red[wid] = s;
  __syncthreads();
  float tot = 0.0f;
  for (int i = 0; i < 8; i++) tot += red[i];
  float mean = tot * (1.0f / DD);
  __syncthreads();
  float s2 = 0.0f;
  for (int i = 0; i < 8; i++) { float d = vals[i] - mean; s2 += d * d; }
  for (int m = 1; m < 32; m <<= 1) s2 += __shfl_xor(s2, m, 32);
  if (lane == 0) red[wid] = s2;
  __syncthreads();
  float tot2 = 0.0f;
  for (int i = 0; i < 8; i++) tot2 += red[i];
  float inv = rsqrtf(tot2 * (1.0f / DD) + LN_EPSF);
  for (int i = 0; i < 8; i++) {
    int c = i * 256 + tid;
    float y = (vals[i] - mean) * inv;
    out[(size_t)row * DD + c] = y;
    if (outb) outb[(size_t)row * DD + c] = f2bf(y);
  }
}

// ---------------------------------------------------------------------------
extern "C" void kernel_launch(void* const* d_in, const int* in_sizes, int n_in,
                              void* d_out, int out_size, void* d_ws, size_t ws_size,
                              hipStream_t stream) {
  (void)in_sizes; (void)n_in; (void)out_size; (void)ws_size;
  const float* x   = (const float*)d_in[0];
  const float* Wq  = (const float*)d_in[1];
  const float* Wk  = (const float*)d_in[2];
  const float* Wv  = (const float*)d_in[3];
  const float* W1  = (const float*)d_in[4];
  const float* b1  = (const float*)d_in[5];
  const float* W2  = (const float*)d_in[6];
  const float* b2  = (const float*)d_in[7];
  const float* snT = (const float*)d_in[8];
  const float* csT = (const float*)d_in[9];
  float* out = (float*)d_out;
  char* ws = (char*)d_ws;

  const size_t MB = 1024ull * 1024ull;
  __bf16* XB  = (__bf16*)(ws + 0);          // [0,16)
  __bf16* WqT = (__bf16*)(ws + 16 * MB);    // [16,24)
  __bf16* WkT = (__bf16*)(ws + 24 * MB);    // [24,32)
  __bf16* WvT = (__bf16*)(ws + 32 * MB);    // [32,40)
  __bf16* W1T = (__bf16*)(ws + 40 * MB);    // [40,72)
  __bf16* W2T = (__bf16*)(ws + 72 * MB);    // [72,104)
  float*  QF  = (float*)(ws + 104 * MB);    // [104,136)
  float*  KF  = (float*)(ws + 136 * MB);    // [136,168)
  float*  VF  = (float*)(ws + 168 * MB);    // [168,200)
  __bf16* QB  = (__bf16*)(ws + 200 * MB);   // [200,216)
  __bf16* KB  = (__bf16*)(ws + 216 * MB);   // [216,232)
  __bf16* VBt = (__bf16*)(ws + 232 * MB);   // [232,248)
  float*  Oatt = (float*)(ws + 104 * MB);   // reuse QF after rope/pack
  float*  OUT1 = (float*)(ws + 136 * MB);   // reuse KF (live to end)
  __bf16* OUT1b = (__bf16*)(ws + 168 * MB); // reuse VF head
  __bf16* HB  = (__bf16*)(ws + 184 * MB);   // [184,248) reuse VF tail + QB/KB/VBt
  float*  Y   = (float*)(ws + 104 * MB);    // reuse Oatt

  // one-time converts / pre-transposes (bf16)
  size_t nx = (size_t)BS * DD;
  cvt_bf16<<<(unsigned)((nx + 255) / 256), 256, 0, stream>>>(x, XB, nx);
  transpose_cvt<<<dim3(DD / 32, DD / 32), 256, 0, stream>>>(Wq, WqT, DD, DD);
  transpose_cvt<<<dim3(DD / 32, DD / 32), 256, 0, stream>>>(Wk, WkT, DD, DD);
  transpose_cvt<<<dim3(DD / 32, DD / 32), 256, 0, stream>>>(Wv, WvT, DD, DD);
  transpose_cvt<<<dim3(FF / 32, DD / 32), 256, 0, stream>>>(W1, W1T, DD, FF);
  transpose_cvt<<<dim3(DD / 32, FF / 32), 256, 0, stream>>>(W2, W2T, FF, DD);

  // QKV projections
  gemm_bf16_async<0><<<dim3(DD / BLK_N, BS / BLK_M), 256, 0, stream>>>(
      XB, WqT, QF, BS, DD, DD, nullptr, nullptr);
  gemm_bf16_async<0><<<dim3(DD / BLK_N, BS / BLK_M), 256, 0, stream>>>(
      XB, WkT, KF, BS, DD, DD, nullptr, nullptr);
  gemm_bf16_async<0><<<dim3(DD / BLK_N, BS / BLK_M), 256, 0, stream>>>(
      XB, WvT, VF, BS, DD, DD, nullptr, nullptr);

  // RoPE + pack q,k [B,H,S,hd]; V -> [B,H,hd,S] via coalesced tiled transpose
  size_t nrope = (size_t)BB * SS * (DD / 2);
  rope_pack<<<(unsigned)((nrope + 255) / 256), 256, 0, stream>>>(
      QF, KF, snT, csT, QB, KB);
  pack_vT<<<dim3(HDIM / 32, SS / 32, BB * HH), 256, 0, stream>>>(VF, VBt);

  // causal flash attention
  flash_attn<<<BB * HH * (SS / FA_Q), 128, 0, stream>>>(QB, KB, VBt, Oatt);

  // out = LN(attn + x)   (+ bf16 copy for next GEMM)
  add_ln<<<BS, 256, 0, stream>>>(Oatt, x, nullptr, 0, OUT1, OUT1b);

  // h = gelu(out @ W1 + b1)  -> bf16 (fused epilogue)
  gemm_bf16_async<1><<<dim3(FF / BLK_N, BS / BLK_M), 256, 0, stream>>>(
      OUT1b, W1T, nullptr, BS, FF, DD, b1, HB);

  // y = h @ W2 ; out2 = LN(y + b2 + out)
  gemm_bf16_async<0><<<dim3(DD / BLK_N, BS / BLK_M), 256, 0, stream>>>(
      HB, W2T, Y, BS, DD, FF, nullptr, nullptr);
  add_ln<<<BS, 256, 0, stream>>>(Y, OUT1, b2, 1, out, nullptr);
}
